// TransformerEncoder_29858612641944
// MI455X (gfx1250) — matrix operand order repaired and checked by reading, not compiled, not run on hardware
//
#include <hip/hip_runtime.h>
#include <math.h>
#include <stdint.h>

// Problem constants (match reference)
#define L_   512
#define B_   8
#define E_   1024
#define H_   16
#define NL_  2
#define HD_  64
#define DFF_ 4096
#define S_   L_
#define MB_  (L_*B_)   // 4096 token rows

// Workspace requirement: ~420 MiB (see layout in kernel_launch).

typedef __attribute__((ext_vector_type(16))) __bf16 v16bf;
typedef __attribute__((ext_vector_type(8)))  float  v8f;

// ---- CDNA5 async global->LDS path (guarded; falls back to VGPR copies) ----
#if defined(__gfx1250__) && __has_builtin(__builtin_amdgcn_global_load_async_to_lds_b128)
#define HAVE_ASYNC 1
#else
#define HAVE_ASYNC 0
#endif

#if HAVE_ASYNC
// Builtin signature (from clang diagnostic): param0 = v4i addrspace(1)*,
// param1 = v4i addrspace(3)*, then imm offset, imm cpol.
typedef int v4i_ __attribute__((vector_size(16)));
typedef __attribute__((address_space(1))) v4i_ gv4i;
typedef __attribute__((address_space(3))) v4i_ lv4i;
#if __has_builtin(__builtin_amdgcn_s_wait_asynccnt)
#define ASYNC_WAIT() __builtin_amdgcn_s_wait_asynccnt(0)
#else
#define ASYNC_WAIT() asm volatile("s_wait_asynccnt 0" ::: "memory")
#endif
#define G2L_B128(g, l, off)                                         \
  __builtin_amdgcn_global_load_async_to_lds_b128(                   \
      (gv4i*)(uintptr_t)(g),                                        \
      (lv4i*)(uint32_t)(uintptr_t)(l), (off), 0)
#else
#define ASYNC_WAIT()
#endif

__device__ __forceinline__ unsigned short f2bf(float f) {
  unsigned u = __builtin_bit_cast(unsigned, f);
  u += 0x7FFFu + ((u >> 16) & 1u);            // round-to-nearest-even
  return (unsigned short)(u >> 16);
}
__device__ __forceinline__ float b2f(unsigned short h) {
  unsigned u = (unsigned)h << 16;
  return __builtin_bit_cast(float, u);
}

// ---------------------------------------------------------------------------
// Strided-batched bf16 GEMM:  C[z] = A[z] (MxK bf16, lda) * Bsrc[z]^T + bias
// Bsrc stored [N][K] row-major (ldb) -- ALL uses are "transB" by construction.
// accumulate: C(f32) += result ; act==1: exact-erf GELU ; out_bf16: C is u16.
// Block tile 128x64, BK=32, 8 waves (4x2), wave tile 32x32 -> 4 WMMAs/K-step.
// Double-buffered LDS; tiles staged with async b128 copies when available.
// Requires M%128==0, N%64==0, K%32==0 (true for all calls here).
// ---------------------------------------------------------------------------
__global__ __launch_bounds__(256)
void gemm_bf16_wmma(const unsigned short* __restrict__ A,
                    const unsigned short* __restrict__ B,
                    void* __restrict__ Cv, const float* __restrict__ bias,
                    int K, int lda, int ldb, int ldc,
                    long long sA, long long sB, long long sC,
                    int accumulate, int act, int out_bf16)
{
  // row stride 40 halves = 80B: 16B-aligned chunks + conflict-free frag reads
  __shared__ __align__(16) unsigned short As[2][128][40];
  __shared__ __align__(16) unsigned short Bs[2][64][40];

  const int tid  = threadIdx.x;
  const int lane = tid & 31;
  const int wave = tid >> 5;
  const int wm   = wave >> 1;          // 0..3 -> 32-row slab
  const int wn   = wave & 1;           // 0..1 -> 32-col slab
  const int bm   = blockIdx.y * 128;
  const int bn   = blockIdx.x * 64;
  const long long z = blockIdx.z;
  const unsigned short* Ab = A + z * sA;
  const unsigned short* Bb = B + z * sB;

  const int lm = lane & 15;            // M-row / N-col within fragment
  const int kh = lane >> 4;            // K-half selector

  // staging coords: A 128x32 (16 bf16/thread), B 64x32 (8 bf16/thread)
  const int ar_ = tid >> 1, ak_ = (tid & 1) * 16;
  const int br_ = tid >> 2, bk_ = (tid & 3) * 8;

  auto stage = [&](int kt, int buf) {
    const unsigned short* ga = Ab + (long long)(bm + ar_) * lda + kt + ak_;
    const unsigned short* gb = Bb + (long long)(bn + br_) * ldb + kt + bk_;
#if HAVE_ASYNC
    G2L_B128(ga, &As[buf][ar_][ak_], 0);
    G2L_B128(ga, &As[buf][ar_][ak_], 16);   // inst offset applies to both addrs
    G2L_B128(gb, &Bs[buf][br_][bk_], 0);
#else
    const uint4* g4 = (const uint4*)ga;
    uint4 d0 = g4[0], d1 = g4[1];
    *(uint4*)&As[buf][ar_][ak_]     = d0;
    *(uint4*)&As[buf][ar_][ak_ + 8] = d1;
    *(uint4*)&Bs[buf][br_][bk_]     = *(const uint4*)gb;
#endif
  };

  v8f c00 = {}, c01 = {}, c10 = {}, c11 = {};
  stage(0, 0);
  ASYNC_WAIT();
  __syncthreads();
  int cur = 0;

  for (int kt = 0; kt < K; kt += 32) {
    if (kt + 32 < K) stage(kt + 32, cur ^ 1);   // overlap DMA with WMMA

    union Frag { v16bf v; unsigned u[8]; } a0, a1, b0, b1;
    const unsigned* ar0 = (const unsigned*)&As[cur][32 * wm + lm][0];
    const unsigned* ar1 = (const unsigned*)&As[cur][32 * wm + 16 + lm][0];
    const unsigned* br0 = (const unsigned*)&Bs[cur][32 * wn + lm][0];
    const unsigned* br1 = (const unsigned*)&Bs[cur][32 * wn + 16 + lm][0];
    #pragma unroll
    for (int j = 0; j < 4; ++j) {
      a0.u[j] = ar0[4 * kh + j];  a0.u[4 + j] = ar0[8 + 4 * kh + j];
      a1.u[j] = ar1[4 * kh + j];  a1.u[4 + j] = ar1[8 + 4 * kh + j];
    }
    #pragma unroll
    for (int j = 0; j < 4; ++j) { b0.u[j] = br0[4 * kh + j]; b0.u[4 + j] = br0[8 + 4 * kh + j]; b1.u[j] = br1[4 * kh + j]; b1.u[4 + j] = br1[8 + 4 * kh + j]; }

    c00 = __builtin_amdgcn_wmma_f32_16x16x32_bf16(false, a0.v, false, b0.v, (short)0, c00, false, false);
    c01 = __builtin_amdgcn_wmma_f32_16x16x32_bf16(false, a0.v, false, b1.v, (short)0, c01, false, false);
    c10 = __builtin_amdgcn_wmma_f32_16x16x32_bf16(false, a1.v, false, b0.v, (short)0, c10, false, false);
    c11 = __builtin_amdgcn_wmma_f32_16x16x32_bf16(false, a1.v, false, b1.v, (short)0, c11, false, false);

    ASYNC_WAIT();
    __syncthreads();
    cur ^= 1;
  }

  float*          Cf = (float*)Cv + z * sC;
  unsigned short* Ch = (unsigned short*)Cv + z * sC;
  const int row0 = bm + 32 * wm + 8 * kh;
  v8f cc[2][2] = {{c00, c01}, {c10, c11}};
  #pragma unroll
  for (int f = 0; f < 2; ++f) {
    #pragma unroll
    for (int t = 0; t < 2; ++t) {
      const int col = bn + 32 * wn + 16 * t + lm;
      const float bv = bias ? bias[col] : 0.0f;
      #pragma unroll
      for (int j = 0; j < 8; ++j) {
        const long long idx = (long long)(row0 + 16 * f + j) * ldc + col;
        float r = cc[f][t][j] + bv;
        if (accumulate) r += Cf[idx];
        if (act == 1)   r = 0.5f * r * (1.0f + erff(r * 0.70710678118654752f));
        if (out_bf16) Ch[idx] = f2bf(r); else Cf[idx] = r;
      }
    }
  }
}

// ---------------------------------------------------------------------------
// Small kernels
// ---------------------------------------------------------------------------
__device__ __forceinline__ float block_sum256(float v, float* red) {
  for (int o = 16; o; o >>= 1) v += __shfl_xor(v, o);
  if ((threadIdx.x & 31) == 0) red[threadIdx.x >> 5] = v;
  __syncthreads();
  float s = 0.0f;
  #pragma unroll
  for (int i = 0; i < 8; ++i) s += red[i];
  __syncthreads();
  return s;
}

__global__ void cvtf2b(const float* __restrict__ a, unsigned short* __restrict__ o, long long n) {
  long long i = (long long)blockIdx.x * blockDim.x + threadIdx.x;
  if (i < n) o[i] = f2bf(a[i]);
}

__global__ void copy_cvt(const float* __restrict__ a, float* __restrict__ xo,
                         unsigned short* __restrict__ xb, long long n) {
  long long i = (long long)blockIdx.x * blockDim.x + threadIdx.x;
  if (i < n) { float v = a[i]; xo[i] = v; xb[i] = f2bf(v); }
}

// rel = pos + typ, stored [L,S,HD] and transposed [L,HD,S] (both bf16)
__global__ __launch_bounds__(256) void build_rel(const float* __restrict__ pos,
                                                 const float* __restrict__ typ,
                                                 unsigned short* __restrict__ rel,
                                                 unsigned short* __restrict__ relT) {
  const long long i = (long long)blockIdx.x * 256 + threadIdx.x;  // L*S*HD
  const int d = (int)(i & (HD_ - 1));
  const int s = (int)((i >> 6) & (S_ - 1));
  const int l = (int)(i >> 15);
  const unsigned short r = f2bf(pos[i] + typ[i]);
  rel[i] = r;
  relT[((long long)l * HD_ + d) * S_ + s] = r;
}

// qkv(bf16,[L,B,3E]) -> q(scaled),k [B,H,L,HD] ; v transposed [B,H,HD,L]
__global__ __launch_bounds__(256) void repack_qkv_k(const unsigned short* __restrict__ qkv,
                                                    unsigned short* __restrict__ q,
                                                    unsigned short* __restrict__ k,
                                                    unsigned short* __restrict__ vT) {
  const int i  = blockIdx.x * 256 + threadIdx.x;   // B*H*L*HD
  const int d  = i & (HD_ - 1);
  const int l  = (i >> 6) & (L_ - 1);
  const int bh = i >> 15;
  const int h  = bh & (H_ - 1);
  const int b  = bh >> 4;
  const unsigned short* p = qkv + ((long long)l * B_ + b) * (3 * E_) + h * HD_ + d;
  q[i] = f2bf(b2f(p[0]) * 0.125f);                 // 1/sqrt(HD)
  k[i] = p[E_];
  vT[((long long)bh * HD_ + d) * L_ + l] = p[2 * E_];
}

// attn(f32,[B,H,L,HD]) -> y0(bf16,[L,B,E])
__global__ __launch_bounds__(256) void repack_attn_k(const float* __restrict__ attn,
                                                     unsigned short* __restrict__ y0) {
  const int i = blockIdx.x * 256 + threadIdx.x;    // L*B*E
  const int e = i & (E_ - 1);
  const int b = (i >> 10) & (B_ - 1);
  const int l = i >> 13;
  const int h = e >> 6, d = e & 63;
  y0[i] = f2bf(attn[(((long long)(b * H_ + h)) * L_ + l) * HD_ + d]);
}

// softmax over rows of 512 (f32 in) -> bf16 probabilities out
__global__ __launch_bounds__(256) void softmax_rows(const float* __restrict__ sc,
                                                    unsigned short* __restrict__ w) {
  __shared__ float red[8];
  const float* row = sc + (long long)blockIdx.x * S_;
  unsigned short* wr = w + (long long)blockIdx.x * S_;
  const int t = threadIdx.x;
  float a = row[t], b = row[t + 256];
  float m = fmaxf(a, b);
  for (int o = 16; o; o >>= 1) m = fmaxf(m, __shfl_xor(m, o));
  if ((t & 31) == 0) red[t >> 5] = m;
  __syncthreads();
  float gm = red[0];
  #pragma unroll
  for (int i = 1; i < 8; ++i) gm = fmaxf(gm, red[i]);
  __syncthreads();
  const float ea = __expf(a - gm), eb = __expf(b - gm);
  float s = ea + eb;
  for (int o = 16; o; o >>= 1) s += __shfl_xor(s, o);
  if ((t & 31) == 0) red[t >> 5] = s;
  __syncthreads();
  float tot = 0.0f;
  #pragma unroll
  for (int i = 0; i < 8; ++i) tot += red[i];
  const float inv = 1.0f / tot;
  wr[t]       = f2bf(ea * inv);
  wr[t + 256] = f2bf(eb * inv);
}

// dst = LayerNorm(x + y)*g + be ; also writes bf16 copy for next GEMM
__global__ __launch_bounds__(256) void add_layernorm(const float* __restrict__ x,
                                                     const float* __restrict__ y,
                                                     const float* __restrict__ g,
                                                     const float* __restrict__ be,
                                                     float* __restrict__ dst,
                                                     unsigned short* __restrict__ dstb) {
  __shared__ float red[8];
  const long long base = (long long)blockIdx.x * E_;
  const int t = threadIdx.x;
  float v[4]; float s = 0.0f;
  #pragma unroll
  for (int i = 0; i < 4; ++i) { const int c = t + 256 * i; v[i] = x[base + c] + y[base + c]; s += v[i]; }
  s = block_sum256(s, red);
  const float mean = s * (1.0f / E_);
  float q = 0.0f;
  #pragma unroll
  for (int i = 0; i < 4; ++i) { const float d = v[i] - mean; q += d * d; }
  q = block_sum256(q, red);
  const float inv = rsqrtf(q * (1.0f / E_) + 1e-5f);
  #pragma unroll
  for (int i = 0; i < 4; ++i) {
    const int c = t + 256 * i;
    const float r = (v[i] - mean) * inv * g[c] + be[c];
    dst[base + c]  = r;
    dstb[base + c] = f2bf(r);
  }
}

// ---------------------------------------------------------------------------
static void gemm(hipStream_t st, const unsigned short* A, const unsigned short* Bp,
                 void* C, const float* bias, int M, int N, int K,
                 int lda, int ldb, int ldc,
                 long long sA, long long sB, long long sC, int batch,
                 int acc, int act, int obf) {
  dim3 grid(N / 64, M / 128, batch);
  gemm_bf16_wmma<<<grid, dim3(256), 0, st>>>(A, Bp, C, bias, K, lda, ldb, ldc,
                                             sA, sB, sC, acc, act, obf);
}

extern "C" void kernel_launch(void* const* d_in, const int* in_sizes, int n_in,
                              void* d_out, int out_size, void* d_ws, size_t ws_size,
                              hipStream_t stream) {
  (void)in_sizes; (void)n_in; (void)out_size; (void)ws_size;
  const float* src = (const float*)d_in[0];
  const float* pos = (const float*)d_in[1];
  const float* typ = (const float*)d_in[2];
  const float* Wi  = (const float*)d_in[3];
  const float* bi  = (const float*)d_in[4];
  const float* Wo  = (const float*)d_in[5];
  const float* bo  = (const float*)d_in[6];
  const float* W1  = (const float*)d_in[7];
  const float* b1  = (const float*)d_in[8];
  const float* W2  = (const float*)d_in[9];
  const float* b2  = (const float*)d_in[10];
  const float* g1  = (const float*)d_in[11];
  const float* be1 = (const float*)d_in[12];
  const float* g2  = (const float*)d_in[13];
  const float* be2 = (const float*)d_in[14];
  float* out = (float*)d_out;

  // ---- workspace layout (256B-aligned bump allocator), total ~420 MiB ----
  char* w8 = (char*)d_ws;
  auto alloc = [&](long long bytes) -> void* {
    void* p = (void*)w8; w8 += (bytes + 255) & ~255LL; return p;
  };
  const long long NREL = (long long)L_ * S_ * HD_;
  const long long NX   = (long long)MB_ * E_;
  unsigned short* rel_bf  = (unsigned short*)alloc(2 * NREL);             // 32M
  unsigned short* relT_bf = (unsigned short*)alloc(2 * NREL);             // 32M
  float*          x       = (float*)alloc(4 * NX);                        // 16M
  unsigned short* x_bf    = (unsigned short*)alloc(2 * NX);               //  8M
  float*          y       = (float*)alloc(4 * NX);                        // 16M
  unsigned short* qkv_bf  = (unsigned short*)alloc(2 * (long long)MB_ * 3 * E_); // 24M
  unsigned short* q_bf    = (unsigned short*)alloc(2 * NX);               //  8M
  unsigned short* k_bf    = (unsigned short*)alloc(2 * NX);               //  8M
  unsigned short* vT_bf   = (unsigned short*)alloc(2 * NX);               //  8M
  float*          attn    = (float*)alloc(4 * NX);                        // 16M
  unsigned short* y0_bf   = (unsigned short*)alloc(2 * NX);               //  8M
  unsigned short* w_bf    = (unsigned short*)alloc(2 * (long long)B_ * H_ * L_ * S_); // 64M
  float*          scores  = (float*)alloc(4 * (long long)B_ * H_ * L_ * S_);          // 128M
  unsigned short* hidden  = (unsigned short*)scores;  // bf16, alias: lifetimes disjoint
  unsigned short* Wi_bf = (unsigned short*)alloc(2LL * NL_ * 3 * E_ * E_);
  unsigned short* Wo_bf = (unsigned short*)alloc(2LL * NL_ * E_ * E_);
  unsigned short* W1_bf = (unsigned short*)alloc(2LL * NL_ * DFF_ * E_);
  unsigned short* W2_bf = (unsigned short*)alloc(2LL * NL_ * E_ * DFF_);

  // ---- one-time prep ----
  cvtf2b<<<(unsigned)((NL_ * 3LL * E_ * E_ + 255) / 256), 256, 0, stream>>>(Wi, Wi_bf, NL_ * 3LL * E_ * E_);
  cvtf2b<<<(unsigned)((NL_ * 1LL * E_ * E_ + 255) / 256), 256, 0, stream>>>(Wo, Wo_bf, NL_ * 1LL * E_ * E_);
  cvtf2b<<<(unsigned)((NL_ * 1LL * DFF_ * E_ + 255) / 256), 256, 0, stream>>>(W1, W1_bf, NL_ * 1LL * DFF_ * E_);
  cvtf2b<<<(unsigned)((NL_ * 1LL * E_ * DFF_ + 255) / 256), 256, 0, stream>>>(W2, W2_bf, NL_ * 1LL * E_ * DFF_);
  build_rel<<<(unsigned)(NREL / 256), 256, 0, stream>>>(pos, typ, rel_bf, relT_bf);
  copy_cvt<<<(unsigned)((NX + 255) / 256), 256, 0, stream>>>(src, x, x_bf, NX);

  for (int i = 0; i < NL_; ++i) {
    const unsigned short* Wi_i = Wi_bf + (long long)i * 3 * E_ * E_;
    const unsigned short* Wo_i = Wo_bf + (long long)i * E_ * E_;
    const unsigned short* W1_i = W1_bf + (long long)i * DFF_ * E_;
    const unsigned short* W2_i = W2_bf + (long long)i * E_ * DFF_;
    const float* bi_i  = bi  + (long long)i * 3 * E_;
    const float* bo_i  = bo  + (long long)i * E_;
    const float* b1_i  = b1  + (long long)i * DFF_;
    const float* b2_i  = b2  + (long long)i * E_;
    const float* g1_i  = g1  + (long long)i * E_;
    const float* be1_i = be1 + (long long)i * E_;
    const float* g2_i  = g2  + (long long)i * E_;
    const float* be2_i = be2 + (long long)i * E_;

    // qkv = x @ Wi^T + bi   -> bf16 [L*B, 3E]
    gemm(stream, x_bf, Wi_i, qkv_bf, bi_i, MB_, 3 * E_, E_, E_, E_, 3 * E_,
         0, 0, 0, 1, 0, 0, /*bf16 out*/1);
    repack_qkv_k<<<MB_ * E_ / 256, 256, 0, stream>>>(qkv_bf, q_bf, k_bf, vT_bf);

    // scores[bh] = q[bh] @ k[bh]^T                 (batch B*H)
    gemm(stream, q_bf, k_bf, scores, nullptr, L_, S_, HD_, HD_, HD_, S_,
         (long long)L_ * HD_, (long long)L_ * HD_, (long long)L_ * S_, B_ * H_, 0, 0, 0);
    // scores[:,l,:] += q[:,l,:] @ rel[l]^T         (batch l)
    gemm(stream, q_bf, rel_bf, scores, nullptr, B_ * H_, S_, HD_,
         L_ * HD_, HD_, L_ * S_,
         (long long)HD_, (long long)S_ * HD_, (long long)S_, L_, /*acc*/1, 0, 0);

    softmax_rows<<<B_ * H_ * L_, 256, 0, stream>>>(scores, w_bf);

    // attn[bh] = w[bh] @ (vT[bh])^T                (batch B*H)
    gemm(stream, w_bf, vT_bf, attn, nullptr, L_, HD_, S_, S_, S_, HD_,
         (long long)L_ * S_, (long long)HD_ * S_, (long long)L_ * HD_, B_ * H_, 0, 0, 0);
    // attn[:,l,:] += w[:,l,:] @ (relT[l])^T        (batch l)
    gemm(stream, w_bf, relT_bf, attn, nullptr, B_ * H_, HD_, S_,
         L_ * S_, S_, L_ * HD_,
         (long long)S_, (long long)HD_ * S_, (long long)HD_, L_, /*acc*/1, 0, 0);

    repack_attn_k<<<MB_ * E_ / 256, 256, 0, stream>>>(attn, y0_bf);

    // y = y0 @ Wo^T + bo
    gemm(stream, y0_bf, Wo_i, y, bo_i, MB_, E_, E_, E_, E_, E_, 0, 0, 0, 1, 0, 0, 0);
    add_layernorm<<<MB_, 256, 0, stream>>>(x, y, g1_i, be1_i, x, x_bf);

    // hidden = gelu(x @ W1^T + b1)  -> bf16
    gemm(stream, x_bf, W1_i, hidden, b1_i, MB_, DFF_, E_, E_, E_, DFF_,
         0, 0, 0, 1, 0, /*gelu*/1, /*bf16 out*/1);
    // y = hidden @ W2^T + b2
    gemm(stream, hidden, W2_i, y, b2_i, MB_, E_, DFF_, DFF_, DFF_, E_, 0, 0, 0, 1, 0, 0, 0);

    float* dst = (i == NL_ - 1) ? out : x;
    add_layernorm<<<MB_, 256, 0, stream>>>(x, y, g2_i, be2_i, dst, x_bf);
  }
}
